// DictionaryConv1D_56573309224026
// MI455X (gfx1250) — matrix operand order repaired and checked by reading, not compiled, not run on hardware
//
#include <hip/hip_runtime.h>

// Problem constants (from reference)
#define NB    32          // batch
#define CIN   64          // input channels
#define LEN   8192        // sequence length
#define NATOM 64          // dictionary atoms A
#define COUT  256         // hidden channels
#define KW    9           // conv kernel width
#define PADW  4           // K//2
#define LT    128         // L tile per workgroup
#define XW    (LT + 2*PADW)   // 136: haloed x tile width (j index)
#define LDR   72          // padded row stride for transposed LDS tiles (16B-aligned, bank-friendly)

// ws layout (bf16): dictT [9][64][64], then w1b [256][64], then w2b [64][256]
#define DICTT_ELEMS (KW * NATOM * CIN)      // 36864
#define W1_ELEMS    (COUT * NATOM)          // 16384
#define W2_ELEMS    (CIN * COUT)            // 16384
#define WS_ELEMS    (DICTT_ELEMS + W1_ELEMS + W2_ELEMS)   // 69632

typedef __bf16 bf16_t;
typedef __attribute__((ext_vector_type(16))) __bf16 v16bf;
typedef __attribute__((ext_vector_type(8)))  float  v8f;

union BF16x16 { v16bf v; uint4 q[2]; };

__device__ __forceinline__ bf16_t f2bf(float f) {
  unsigned u = __builtin_bit_cast(unsigned, f);
  unsigned r = (u + 0x7FFFu + ((u >> 16) & 1u)) >> 16;   // RNE
  unsigned short s = (unsigned short)r;
  return __builtin_bit_cast(bf16_t, s);
}

// Pack two floats into packed-bf16 dword with RNE: [15:0]=bf(a), [31:16]=bf(b).
// Rounding stays in 32-bit regs; final pack is one v_perm_b32.
__device__ __forceinline__ unsigned pk_bf2(float a, float b) {
  unsigned ua = __builtin_bit_cast(unsigned, a);
  unsigned ub = __builtin_bit_cast(unsigned, b);
  ua += 0x7FFFu + ((ua >> 16) & 1u);
  ub += 0x7FFFu + ((ub >> 16) & 1u);
  return __builtin_amdgcn_perm(ub, ua, 0x07060302);      // {ub[31:16], ua[31:16]}
}

__device__ __forceinline__ v8f wmma_bf16(v16bf a, v16bf b, v8f c) {
  return __builtin_amdgcn_wmma_f32_16x16x32_bf16(false, a, false, b, (short)0, c,
                                                 false, false);
}

// A-fragment (16x32) from bf16 row-major W[M][ldk] in global memory.
// Layout used here (contiguous k, same as B): M = mbase+(lane&15) for all lanes; elements 0..15 hold
// K=16h+{0..15}  ->  one contiguous 32B run (two 16B loads).
__device__ __forceinline__ v16bf load_a_bf(const bf16_t* __restrict__ W,
                                           int mbase, int kbase, int ldk, int lane) {
  const int h = (lane >> 4) & 1;
  const bf16_t* rp = W + (size_t)(mbase + (lane & 15)) * ldk + kbase + 16 * h;
  BF16x16 r;
  r.q[0] = *(const uint4*)(rp);          // K = kbase+16h+{0..7}
  r.q[1] = *(const uint4*)(rp + 8);      // K = kbase+16h+{8..15}
  return r.v;
}

// B-fragment (32x16) from transposed bf16 LDS tile T[col][LDR].
// Layout here (contiguous k, as A): N = col+(lane&15); elements 0..15 are K = kbase+16h+{0..15}
// -> one contiguous 32B run per lane = two ds_load_b128.
__device__ __forceinline__ v16bf load_b_lds(const bf16_t* T, int kbase, int col,
                                            int lane) {
  const int h = (lane >> 4) & 1;
  const bf16_t* cp = T + (size_t)(col + (lane & 15)) * LDR + kbase + 16 * h;
  BF16x16 r;
  r.q[0] = *(const uint4*)(cp);
  r.q[1] = *(const uint4*)(cp + 8);
  return r.v;
}

// C/D-fragment store to transposed bf16 LDS tile T[col][LDR]:
// lane owns one column; rows M = mbase+8h+{0..7} contiguous -> one 16B store.
__device__ __forceinline__ void store_c_t(bf16_t* T, v8f acc, int mbase, int col,
                                          int lane, bool relu) {
  const int h = (lane >> 4) & 1;
  bf16_t* p = T + (size_t)(col + (lane & 15)) * LDR + mbase + 8 * h;
  uint4 q;
  if (relu) {
#pragma unroll
    for (int r = 0; r < 8; ++r) acc[r] = fmaxf(acc[r], 0.0f);
  }
  q.x = pk_bf2(acc[0], acc[1]);
  q.y = pk_bf2(acc[2], acc[3]);
  q.z = pk_bf2(acc[4], acc[5]);
  q.w = pk_bf2(acc[6], acc[7]);
  *(uint4*)p = q;
}

__device__ __forceinline__ v8f bias_frag(const float* __restrict__ bias, int mbase,
                                         int lane) {
  const int h = (lane >> 4) & 1;
  v8f c;
#pragma unroll
  for (int r = 0; r < 8; ++r) c[r] = bias[mbase + 8 * h + r];
  return c;
}

// ---------------------------------------------------------------------------
// Prep kernel: convert weights to bf16 in fragment-friendly layouts (runs
// every launch; ~70K elements, negligible).
//   dictT[k][a][c] = dict[a][c][k]   (per-tap 64x64 GEMM A-matrices)
//   w1b  = bf16(w1) [256][64]
//   w2b  = bf16(w2) [64][256]
// ---------------------------------------------------------------------------
__global__ __launch_bounds__(256)
void prep_weights_kernel(const float* __restrict__ dict,
                         const float* __restrict__ w1,
                         const float* __restrict__ w2,
                         bf16_t* __restrict__ ws) {
  const int t = blockIdx.x * 256 + threadIdx.x;
  if (t < DICTT_ELEMS) {
    const int k = t >> 12;          // / (64*64)
    const int a = (t >> 6) & 63;
    const int c = t & 63;
    ws[t] = f2bf(dict[a * (CIN * KW) + c * KW + k]);
  } else if (t < DICTT_ELEMS + W1_ELEMS) {
    const int i = t - DICTT_ELEMS;
    ws[t] = f2bf(w1[i]);
  } else if (t < WS_ELEMS) {
    const int i = t - DICTT_ELEMS - W1_ELEMS;
    ws[t] = f2bf(w2[i]);
  }
}

// ---------------------------------------------------------------------------
// Fused conv + 1x1 + 1x1 kernel. One WG per (batch, 128-wide L tile);
// 8 wave32 waves, each owning a 16-column slab for all three GEMMs.
// ---------------------------------------------------------------------------
__global__ __launch_bounds__(256)
void dict_conv_fused_kernel(const float* __restrict__ x,
                            const bf16_t* __restrict__ ws,
                            const float* __restrict__ b1,
                            const float* __restrict__ b2,
                            float* __restrict__ out) {
  __shared__ __align__(16) bf16_t sXT[XW * LDR];   // x tile, [j][c], c-padded
  __shared__ __align__(16) bf16_t sS[LT * LDR];    // sparse maps, [col][atom]
  __shared__ __align__(16) bf16_t sF[LT * LDR];    // feature chunk, [col][feat]

  const bf16_t* __restrict__ dictT = ws;                          // [9][64][64]
  const bf16_t* __restrict__ w1b   = ws + DICTT_ELEMS;            // [256][64]
  const bf16_t* __restrict__ w2b   = ws + DICTT_ELEMS + W1_ELEMS; // [64][256]

  const int wg   = blockIdx.x;
  const int b    = wg / (LEN / LT);
  const int l0   = (wg - b * (LEN / LT)) * LT;
  const int tid  = threadIdx.x;
  const int lane = tid & 31;
  const int wave = tid >> 5;
  const int ncol = wave * 16;                      // this wave's column slab

  // ---- Stage 0: haloed x tile, fp32 -> bf16, transposed [j][c] ------------
  const float* xb = x + (size_t)b * CIN * LEN;
  for (int idx = tid; idx < CIN * XW; idx += 256) {
    const int c = idx / XW;              // global reads coalesced along j
    const int j = idx - c * XW;
    const int gl = l0 - PADW + j;
    float v = 0.0f;
    if (gl >= 0 && gl < LEN) v = xb[(size_t)c * LEN + gl];
    sXT[j * LDR + c] = f2bf(v);
  }
  __syncthreads();
  // After this barrier each wave touches only its own column slab of sS/sF.

  // ---- Stage 1: conv as 9 shifted GEMMs (M=64 atoms, K=64 channels) -------
  {
    v8f acc[4];
#pragma unroll
    for (int m = 0; m < 4; ++m) {
#pragma unroll
      for (int r = 0; r < 8; ++r) acc[m][r] = 0.0f;
    }
    for (int ktap = 0; ktap < KW; ++ktap) {
      const bf16_t* dk = dictT + ktap * (NATOM * CIN);
      const int jbase = ncol + ktap;     // shifted column into haloed tile
#pragma unroll
      for (int kc = 0; kc < 2; ++kc) {
        const v16bf bfr = load_b_lds(sXT, kc * 32, jbase, lane);
#pragma unroll
        for (int m = 0; m < 4; ++m) {
          const v16bf afr = load_a_bf(dk, m * 16, kc * 32, CIN, lane);
          acc[m] = wmma_bf16(afr, bfr, acc[m]);
        }
      }
    }
#pragma unroll
    for (int m = 0; m < 4; ++m)
      store_c_t(sS, acc[m], m * 16, ncol, lane, false);
  }

  // ---- Stages 2+3 fused over 4 chunks of 64 hidden channels ---------------
  v8f acc3[4];
#pragma unroll
  for (int m = 0; m < 4; ++m) acc3[m] = bias_frag(b2, m * 16, lane);

  // sS B-fragments are invariant across the whole stage-2 loop: hoist.
  const v16bf bS0 = load_b_lds(sS, 0,  ncol, lane);
  const v16bf bS1 = load_b_lds(sS, 32, ncol, lane);

  for (int cc = 0; cc < 4; ++cc) {
    // Stage 2: F[cc*64 .. +63] = relu(w1 * S + b1) into sF (own slab only)
#pragma unroll
    for (int m2 = 0; m2 < 4; ++m2) {
      v8f acc = bias_frag(b1, cc * 64 + m2 * 16, lane);
      acc = wmma_bf16(load_a_bf(w1b, cc * 64 + m2 * 16, 0,  NATOM, lane), bS0, acc);
      acc = wmma_bf16(load_a_bf(w1b, cc * 64 + m2 * 16, 32, NATOM, lane), bS1, acc);
      store_c_t(sF, acc, m2 * 16, ncol, lane, true);
    }
    // Stage 3: accumulate w2[:, cc*64 .. +63] * Fchunk
#pragma unroll
    for (int kk = 0; kk < 2; ++kk) {
      const v16bf bfr = load_b_lds(sF, kk * 32, ncol, lane);
#pragma unroll
      for (int m = 0; m < 4; ++m) {
        const v16bf afr = load_a_bf(w2b, m * 16, cc * 64 + kk * 32, COUT, lane);
        acc3[m] = wmma_bf16(afr, bfr, acc3[m]);
      }
    }
  }

  // ---- Epilogue: out[b, M, l0+col] (fp32, coalesced across lanes) ---------
  {
    const int h = (lane >> 4) & 1;
    const int n = l0 + ncol + (lane & 15);
    float* ob = out + (size_t)b * CIN * LEN;
#pragma unroll
    for (int m = 0; m < 4; ++m) {
#pragma unroll
      for (int r = 0; r < 8; ++r) {
        const int M = m * 16 + 8 * h + r;
        ob[(size_t)M * LEN + n] = acc3[m][r];
      }
    }
  }
}

extern "C" void kernel_launch(void* const* d_in, const int* in_sizes, int n_in,
                              void* d_out, int out_size, void* d_ws, size_t ws_size,
                              hipStream_t stream) {
  (void)in_sizes; (void)n_in; (void)out_size; (void)ws_size;
  const float* x    = (const float*)d_in[0];
  const float* dict = (const float*)d_in[1];
  const float* w1   = (const float*)d_in[2];
  const float* b1   = (const float*)d_in[3];
  const float* w2   = (const float*)d_in[4];
  const float* b2   = (const float*)d_in[5];
  float*  out = (float*)d_out;
  bf16_t* ws  = (bf16_t*)d_ws;           // needs WS_ELEMS*2 = ~136 KB scratch

  // 1) convert weights to bf16 fragment layouts (in-order on stream)
  prep_weights_kernel<<<dim3((WS_ELEMS + 255) / 256), dim3(256), 0, stream>>>(
      dict, w1, w2, ws);

  // 2) fused pipeline: 32 batches * 64 L-tiles = 2048 WGs of 8 waves
  dict_conv_fused_kernel<<<dim3(NB * (LEN / LT)), dim3(256), 0, stream>>>(
      x, ws, b1, b2, out);
}
